// VectorQuantizer_10608569221271
// MI455X (gfx1250) — matrix operand order repaired and checked by reading, not compiled, not run on hardware
//
#include <hip/hip_runtime.h>

typedef __bf16 bf16;
typedef __attribute__((ext_vector_type(16))) __bf16 v16bf;
typedef __attribute__((ext_vector_type(8)))  __bf16 v8bf;
typedef __attribute__((ext_vector_type(8)))  float  v8f;

#define K_CODES        1024
#define DIMS           64
#define LDS_STRIDE     72          // padded bf16 elems per codebook row: 144B, 16B aligned
#define WG_THREADS     256
#define ROWS_PER_WAVE  32          // two 16-row WMMA tiles per wave
#define ROWS_PER_WG    (8 * ROWS_PER_WAVE)   // 256 rows per workgroup

__device__ __forceinline__ void fsplit(float x, bf16& h, bf16& l) {
    h = (bf16)x;                    // RNE to bf16
    l = (bf16)(x - (float)h);       // residual, also bf16
}

extern "C" __global__ __launch_bounds__(WG_THREADS)
void vq_argmin_kernel(const float* __restrict__ z_e,
                      const float* __restrict__ cb,
                      float* __restrict__ out, int N) {
    extern __shared__ char smem[];
    bf16*  chi   = (bf16*)smem;                          // [1024][72] hi
    bf16*  clo   = chi + K_CODES * LDS_STRIDE;           // [1024][72] lo
    float* snorm = (float*)(clo + K_CODES * LDS_STRIDE); // [1024] 0.5*||c||^2

    const int tid = threadIdx.x;

    // ---- stage codebook into LDS as bf16 hi/lo (coalesced) ----
    for (int idx = tid; idx < K_CODES * DIMS; idx += WG_THREADS) {
        int c = idx >> 6, d = idx & 63;
        float x = cb[idx];
        bf16 h, l; fsplit(x, h, l);
        chi[c * LDS_STRIDE + d] = h;
        clo[c * LDS_STRIDE + d] = l;
    }
    __syncthreads();
    // ---- per-code half squared norms ----
    for (int c = tid; c < K_CODES; c += WG_THREADS) {
        float s = 0.f;
        #pragma unroll
        for (int d = 0; d < DIMS; ++d) {
            float x = (float)chi[c * LDS_STRIDE + d] + (float)clo[c * LDS_STRIDE + d];
            s += x * x;
        }
        snorm[c] = 0.5f * s;
    }
    __syncthreads();

    const int lane = tid & 31;
    const int wave = tid >> 5;
    const int sub  = lane & 15;     // M index (A/C) or N index (B)
    const int grp  = lane >> 4;     // lane half-group

    float* z_q_out = out;
    float* idx_out = out + (size_t)N * DIMS;

    const int rowBase = blockIdx.x * ROWS_PER_WG + wave * ROWS_PER_WAVE;

    // ---- load two 16x64 A tiles (rows rowBase..+15 and +16..+31), hi/lo split,
    //      packed per CDNA5 A layout: lane sub = M, grp selects K subsets ----
    v16bf ahi[2][2], alo[2][2];     // [tile][kchunk]
    #pragma unroll
    for (int t = 0; t < 2; ++t) {
        const float* zr = z_e + (size_t)(rowBase + 16 * t + sub) * DIMS;
        #pragma unroll
        for (int kc = 0; kc < 2; ++kc) {
            const float4* p0 = (const float4*)(zr + 32 * kc + 8 * grp);       // K = 8g..8g+7
            const float4* p1 = (const float4*)(zr + 32 * kc + 16 + 8 * grp);  // K = 16+8g..
            float4 q0 = p0[0], q1 = p0[1], q2 = p1[0], q3 = p1[1];
            float f[16] = {q0.x,q0.y,q0.z,q0.w, q1.x,q1.y,q1.z,q1.w,
                           q2.x,q2.y,q2.z,q2.w, q3.x,q3.y,q3.z,q3.w};
            #pragma unroll
            for (int e = 0; e < 16; ++e) {
                bf16 h, l; fsplit(f[e], h, l);
                ahi[t][kc][e] = h;
                alo[t][kc][e] = l;
            }
        }
    }

    float best[2][8]; int bestk[2][8];
    #pragma unroll
    for (int t = 0; t < 2; ++t)
        #pragma unroll
        for (int r = 0; r < 8; ++r) { best[t][r] = -3.0e38f; bestk[t][r] = 0; }

    // ---- sweep all 1024 codes, 16 columns per tile; reuse each B load for 2 A tiles ----
    for (int ct = 0; ct < K_CODES / 16; ++ct) {
        const int code = ct * 16 + sub;   // B: column n = sub
        v8f acc0[2] = {{}, {}};           // zh*ch chains
        v8f acc1[2] = {{}, {}};           // correction chains (zl*ch + zh*cl)
        #pragma unroll
        for (int kc = 0; kc < 2; ++kc) {
            const v8bf* bhp = (const v8bf*)(chi + code * LDS_STRIDE + kc * 32 + grp * 8);
            const v8bf* blp = (const v8bf*)(clo + code * LDS_STRIDE + kc * 32 + grp * 8);
            v8bf bh0 = bhp[0], bh1 = bhp[2];
            v8bf bl0 = blp[0], bl1 = blp[2];
            v16bf Bh = __builtin_shufflevector(bh0, bh1, 0,1,2,3,4,5,6,7,8,9,10,11,12,13,14,15);
            v16bf Bl = __builtin_shufflevector(bl0, bl1, 0,1,2,3,4,5,6,7,8,9,10,11,12,13,14,15);
            #pragma unroll
            for (int t = 0; t < 2; ++t) {
                acc0[t] = __builtin_amdgcn_wmma_f32_16x16x32_bf16(false, ahi[t][kc], false, Bh,
                                                                  (short)0, acc0[t], false, false);
                acc1[t] = __builtin_amdgcn_wmma_f32_16x16x32_bf16(false, alo[t][kc], false, Bh,
                                                                  (short)0, acc1[t], false, false);
                acc1[t] = __builtin_amdgcn_wmma_f32_16x16x32_bf16(false, ahi[t][kc], false, Bl,
                                                                  (short)0, acc1[t], false, false);
            }
        }
        const float sn = snorm[code];
        #pragma unroll
        for (int t = 0; t < 2; ++t) {
            #pragma unroll
            for (int r = 0; r < 8; ++r) {
                float v = (acc0[t][r] + acc1[t][r]) - sn;   // = dot - 0.5*||c||^2
                if (v > best[t][r]) { best[t][r] = v; bestk[t][r] = code; }
            }
        }
    }

    #pragma unroll
    for (int t = 0; t < 2; ++t) {
        // ---- reduce across the 16 lanes of each half-group (rows 0-7 / 8-15) ----
        #pragma unroll
        for (int r = 0; r < 8; ++r) {
            float v = best[t][r]; int k = bestk[t][r];
            #pragma unroll
            for (int m = 8; m >= 1; m >>= 1) {
                float ov = __shfl_xor(v, m, 32);
                int   ok = __shfl_xor(k, m, 32);
                if (ov > v || (ov == v && ok < k)) { v = ov; k = ok; }
            }
            best[t][r] = v; bestk[t][r] = k;
        }

        const int tileRow = rowBase + 16 * t;
        // ---- write indices (as float; one lane per row) ----
        #pragma unroll
        for (int r = 0; r < 8; ++r) {
            if (sub == r) idx_out[tileRow + grp * 8 + r] = (float)bestk[t][r];
        }
        // ---- gather z_q rows from the f32 codebook (bit-exact), coalesced float4 ----
        #pragma unroll
        for (int r = 0; r < 8; ++r) {
            const int row = tileRow + grp * 8 + r;
            const int k   = bestk[t][r];
            const float4* src = (const float4*)(cb + (size_t)k * DIMS) + sub;
            float4* dst = (float4*)(z_q_out + (size_t)row * DIMS) + sub;
            *dst = *src;
        }
    }
}

extern "C" void kernel_launch(void* const* d_in, const int* in_sizes, int n_in,
                              void* d_out, int out_size, void* d_ws, size_t ws_size,
                              hipStream_t stream) {
    const float* z_e = (const float*)d_in[0];
    const float* cb  = (const float*)d_in[1];
    float* out = (float*)d_out;
    const int N = in_sizes[0] / DIMS;                 // 262144

    const size_t shmem = (size_t)K_CODES * LDS_STRIDE * sizeof(bf16) * 2
                       + (size_t)K_CODES * sizeof(float);   // ~292 KB (<= 320 KB/WGP)
    const int grid = N / ROWS_PER_WG;                 // 1024 workgroups

    vq_argmin_kernel<<<grid, WG_THREADS, shmem, stream>>>(z_e, cb, out, N);
}
